// GPT_1460288880984
// MI455X (gfx1250) — matrix operand order repaired and checked by reading, not compiled, not run on hardware
//
#include <hip/hip_runtime.h>

// ---------------- model dims (match reference) ----------------
constexpr int V  = 32000;
constexpr int T  = 1024;
constexpr int E  = 1024;
constexpr int H  = 16;
constexpr int L  = 6;
constexpr int D  = 64;          // E / H
constexpr int FF = 4096;        // 4 * E
constexpr int B  = 2;
constexpr int M0 = B * T;       // 2048 rows

// ---------------- vector types ----------------
typedef _Float16     h2    __attribute__((ext_vector_type(2)));
typedef _Float16     h8    __attribute__((ext_vector_type(8)));
typedef _Float16     v16h  __attribute__((ext_vector_type(16)));
typedef float        v8f   __attribute__((ext_vector_type(8)));
typedef float        f4    __attribute__((ext_vector_type(4)));
typedef unsigned int u32x4 __attribute__((ext_vector_type(4)));
typedef int          i32x4 __attribute__((ext_vector_type(4)));
typedef int          i32x8 __attribute__((ext_vector_type(8)));

static __device__ inline v8f zero8() {
    v8f z = {0.f, 0.f, 0.f, 0.f, 0.f, 0.f, 0.f, 0.f};
    return z;
}
static __device__ inline v16h cat16(h8 a, h8 b) {
    return __builtin_shufflevector(a, b, 0, 1, 2, 3, 4, 5, 6, 7,
                                   8, 9, 10, 11, 12, 13, 14, 15);
}

#if defined(__HIP_DEVICE_COMPILE__) && \
    __has_builtin(__builtin_amdgcn_tensor_load_to_lds) && \
    __has_builtin(__builtin_amdgcn_s_wait_tensorcnt)
#define USE_TDM 1
#else
#define USE_TDM 0
#endif

// =================================================================
// Embedding: x[b,t,:] = tok_w[idx[b,t],:] + pos_emb[0,t,:]
// =================================================================
__global__ void embed_kernel(const int* __restrict__ idx,
                             const float* __restrict__ tok_w,
                             const float* __restrict__ pos,
                             float* __restrict__ x) {
    int m   = blockIdx.x;
    int t   = m % T;
    int tok = idx[m];
    const float* tw = tok_w + (size_t)tok * E;
    const float* pw = pos + (size_t)t * E;
    float* xr = x + (size_t)m * E;
    for (int c = threadIdx.x; c < E; c += 256)
        xr[c] = tw[c] + pw[c];
}

// =================================================================
// LayerNorm (fp32 in) -> f16 out.  One block (256 thr) per row.
// =================================================================
__global__ void ln_kernel(const float* __restrict__ x,
                          const float* __restrict__ sc,
                          const float* __restrict__ bs,
                          _Float16* __restrict__ out) {
    int row = blockIdx.x;
    int tid = threadIdx.x;
    const float* xr = x + (size_t)row * E;

    float vals[4];
    float lsum = 0.f, lsq = 0.f;
#pragma unroll
    for (int i = 0; i < 4; i++) {
        float t = xr[tid + i * 256];
        vals[i] = t;
        lsum += t;
        lsq  += t * t;
    }
    __shared__ float r1[256], r2[256];
    r1[tid] = lsum; r2[tid] = lsq;
    __syncthreads();
    for (int off = 128; off > 0; off >>= 1) {
        if (tid < off) { r1[tid] += r1[tid + off]; r2[tid] += r2[tid + off]; }
        __syncthreads();
    }
    float mean = r1[0] * (1.0f / E);
    float var  = r2[0] * (1.0f / E) - mean * mean;
    float inv  = rsqrtf(var + 1e-5f);
    _Float16* orow = out + (size_t)row * E;
#pragma unroll
    for (int i = 0; i < 4; i++) {
        int c = tid + i * 256;
        orow[c] = (_Float16)((vals[i] - mean) * inv * sc[c] + bs[c]);
    }
}

// =================================================================
// WMMA GEMM:  C[M,N] = act( A16[M,K] @ W[K,N] (+bias) (+resid) )
// 128x128 block tile, 8 waves (256 thr), each wave 32x64 (2x4 WMMA
// tiles), K stepped 32.  A tile fetched via Tensor Data Mover (TDM)
// when available (wave 0 issues tensor_load_to_lds, drains with
// s_wait_tensorcnt); W tile staged fp32->f16 *transposed* so B
// fragments are contiguous 16B LDS loads.
// flags: 1=relu, 2=residual(fp32), 4=bias, 8=f16 output
// =================================================================
__global__ __launch_bounds__(256)
void wmma_gemm(const _Float16* __restrict__ A,
               const float* __restrict__ W,
               const float* __restrict__ bias,
               const float* __restrict__ resid,
               float* __restrict__ outF,
               _Float16* __restrict__ outH,
               int M, int N, int K, int flags) {
    __shared__ _Float16 As[128][40];    // 128x32 tile, stride 40 halfs (80B)
    __shared__ _Float16 WsT[128][40];   // transposed W tile: [col][k]

    int tid  = threadIdx.x;
    int lane = tid & 31;
    int wave = tid >> 5;
    int wm   = wave >> 1;      // 0..3  (row group of 32)
    int wn   = wave & 1;       // 0..1  (col group of 64)
    int l15  = lane & 15;
    int hi   = lane >> 4;

    int bm = blockIdx.y * 128;
    int bn = blockIdx.x * 128;

    v8f acc[2][4];
#pragma unroll
    for (int mi = 0; mi < 2; mi++)
#pragma unroll
        for (int ni = 0; ni < 4; ni++) acc[mi][ni] = zero8();

#if USE_TDM
    unsigned lds_base = (unsigned)(unsigned long long)(uintptr_t)&As[0][0];
#endif

    for (int kb = 0; kb < K; kb += 32) {
#if USE_TDM
        // ---- A tile via Tensor Data Mover: 2D tile 128 rows x 32 halfs,
        // data_size=2B, LDS padding 4 DWORDs after every 16 DWORDs (-> 40h).
        if (wave == 0) {
            unsigned long long ga =
                (unsigned long long)(uintptr_t)(A + (size_t)bm * K + kb);
            u32x4 g0;
            g0[0] = 1u;                                   // count=1, user D#
            g0[1] = lds_base;                             // lds_addr
            g0[2] = (unsigned)ga;                         // global_addr lo
            g0[3] = (unsigned)((ga >> 32) & 0x1FFFFFFu) | (2u << 30); // type=2
            i32x8 g1;
            g1[0] = (int)((1u << 16) |       // data_size = 2 bytes
                          (1u << 20) |       // pad_enable
                          (3u << 22) |       // pad_interval: 16 DWORDs
                          (3u << 25));       // pad_amount:   4 DWORDs
            g1[1] = (int)(((unsigned)K & 0xFFFFu) << 16);          // td0 lo
            g1[2] = (int)((((unsigned)K >> 16) & 0xFFFFu) |
                          (((unsigned)M & 0xFFFFu) << 16));        // td0 hi|td1 lo
            g1[3] = (int)(((((unsigned)M >> 16) & 0xFFFFu)) |
                          (32u << 16));                            // tile_dim0=32
            g1[4] = 128;                                           // tile_dim1=128
            g1[5] = (int)(unsigned)K;                              // dim0 stride
            g1[6] = 0;
            g1[7] = 0;
            i32x4 z4 = {0, 0, 0, 0};
#if __clang_major__ >= 23
            i32x8 z8 = {0, 0, 0, 0, 0, 0, 0, 0};
            __builtin_amdgcn_tensor_load_to_lds(g0, g1, z4, z4, z8, 0);
#else
            __builtin_amdgcn_tensor_load_to_lds(g0, g1, z4, z4, 0);
#endif
            __builtin_amdgcn_s_wait_tensorcnt(0);
        }
#else
        // ---- A tile staged synchronously: 2 threads per row, 16 halfs each
        {
            int r = tid >> 1, c = (tid & 1) * 16;
            const _Float16* ap = A + (size_t)(bm + r) * K + kb + c;
            *(h8*)&As[r][c]     = *(const h8*)ap;
            *(h8*)&As[r][c + 8] = *(const h8*)(ap + 8);
        }
#endif
        // ---- W tile fp32 -> f16, stored transposed: WsT[col][k]
        {
            int k  = tid >> 3;            // 0..31
            int c0 = (tid & 7) * 16;      // 0..112
            const float* wp = W + (size_t)(kb + k) * N + bn + c0;
#pragma unroll
            for (int i = 0; i < 16; i += 4) {
                f4 f = *(const f4*)(wp + i);
                WsT[c0 + i + 0][k] = (_Float16)f.x;
                WsT[c0 + i + 1][k] = (_Float16)f.y;
                WsT[c0 + i + 2][k] = (_Float16)f.z;
                WsT[c0 + i + 3][k] = (_Float16)f.w;
            }
        }
        if (kb + 32 < K)
            __builtin_prefetch(W + (size_t)(kb + 32) * N + bn, 0, 0);
        __syncthreads();

        // ---- fragments: contiguous 16B LDS loads + register concat
        v16h af[2], bf[4];
#pragma unroll
        for (int mi = 0; mi < 2; mi++) {
            int row = wm * 32 + mi * 16 + l15;
            af[mi] = cat16(*(const h8*)&As[row][hi * 8],
                           *(const h8*)&As[row][16 + hi * 8]);
        }
#pragma unroll
        for (int ni = 0; ni < 4; ni++) {
            int col = wn * 64 + ni * 16 + l15;
            bf[ni] = cat16(*(const h8*)&WsT[col][hi * 8],
                           *(const h8*)&WsT[col][16 + hi * 8]);
        }
#pragma unroll
        for (int mi = 0; mi < 2; mi++)
#pragma unroll
            for (int ni = 0; ni < 4; ni++)
                acc[mi][ni] = __builtin_amdgcn_wmma_f32_16x16x32_f16(
                    false, af[mi], false, bf[ni], (short)0, acc[mi][ni],
                    false, false);
        __syncthreads();
    }

    // ---- epilogue: bias / residual / relu / f16-or-f32 store
#pragma unroll
    for (int mi = 0; mi < 2; mi++) {
#pragma unroll
        for (int ni = 0; ni < 4; ni++) {
            int n = bn + wn * 64 + ni * 16 + l15;
            float bv = (flags & 4) ? bias[n] : 0.0f;
#pragma unroll
            for (int r = 0; r < 8; r++) {
                int m = bm + wm * 32 + mi * 16 + r + 8 * hi;   // C layout
                size_t off = (size_t)m * N + n;
                float val = acc[mi][ni][r] + bv;
                if (flags & 2) val += resid[off];
                if (flags & 1) val = fmaxf(val, 0.0f);
                if (flags & 8) outH[off] = (_Float16)val;
                else           outF[off] = val;
            }
        }
    }
}

// =================================================================
// Causal attention, flash-style.  One wave per (b, h, 16-query tile).
// q/k/v/o are f16 in [B,T,H*D] layout.  Per 32-key block: 4 WMMAs for
// S = Q K^T (Q/K fragments are contiguous global 16B loads), online
// softmax (width-16 shuffles matching the C layout), P restaged via
// LDS, V tile transposed via LDS, 4 WMMAs for O += P V.
// =================================================================
__global__ __launch_bounds__(32)
void attn_kernel(const _Float16* __restrict__ q,
                 const _Float16* __restrict__ kk,
                 const _Float16* __restrict__ vv,
                 _Float16* __restrict__ o) {
    int qb = blockIdx.x;
    int bh = blockIdx.y;
    int b  = bh >> 4;             // H == 16
    int h  = bh & 15;
    int lane = threadIdx.x;
    int l15  = lane & 15;
    int hi   = lane >> 4;

    __shared__ _Float16 P[16][40];     // probabilities tile (A-major)
    __shared__ _Float16 VT[64][40];    // V tile transposed: [d][key]

    // --- Q A-fragments (contiguous 16B global loads)
    const _Float16* qbase = q + (size_t)(b * T + qb * 16 + l15) * E + h * D;
    v16h qa[2];
#pragma unroll
    for (int part = 0; part < 2; part++)
        qa[part] = cat16(*(const h8*)(qbase + part * 32 + hi * 8),
                         *(const h8*)(qbase + part * 32 + 16 + hi * 8));

    float m_[8], l_[8];
    v8f oacc[4];
#pragma unroll
    for (int r = 0; r < 8; r++) { m_[r] = -1e30f; l_[r] = 0.f; }
#pragma unroll
    for (int nt = 0; nt < 4; nt++) oacc[nt] = zero8();

    const float Csc = 0.18033688f;   // (1/sqrt(64)) * log2(e)
    int nkb = (qb * 16 + 47) >> 5;

    for (int kbi = 0; kbi < nkb; kbi++) {
        int kb = kbi * 32;
        v8f s0 = zero8(), s1 = zero8();

        // --- S = Q K^T : K B-fragments are two 8-half (16B) global runs
#pragma unroll
        for (int tile = 0; tile < 2; tile++) {
            const _Float16* kbase =
                kk + (size_t)(b * T + kb + tile * 16 + l15) * E + h * D;
#pragma unroll
            for (int part = 0; part < 2; part++) {
                v16h bf = cat16(
                    *(const h8*)(kbase + part * 32 + hi * 8),
                    *(const h8*)(kbase + part * 32 + 16 + hi * 8));
                if (tile == 0)
                    s0 = __builtin_amdgcn_wmma_f32_16x16x32_f16(
                        false, qa[part], false, bf, (short)0, s0, false, false);
                else
                    s1 = __builtin_amdgcn_wmma_f32_16x16x32_f16(
                        false, qa[part], false, bf, (short)0, s1, false, false);
            }
        }

        // --- stage V tile transposed: lane loads key row kb+lane (64 halfs)
        {
            const _Float16* vrow =
                vv + (size_t)(b * T + kb + lane) * E + h * D;
#pragma unroll
            for (int i = 0; i < 8; i++) {
                h8 rv = *(const h8*)(vrow + i * 8);
#pragma unroll
                for (int jj = 0; jj < 8; jj++)
                    VT[i * 8 + jj][lane] = rv[jj];
            }
        }

        // --- online softmax update
#pragma unroll
        for (int r = 0; r < 8; r++) {
            int qrow = qb * 16 + r + 8 * hi;
            int key0 = kb + l15;
            int key1 = key0 + 16;
            float x0 = (key0 <= qrow) ? s0[r] * Csc : -1e30f;
            float x1 = (key1 <= qrow) ? s1[r] * Csc : -1e30f;
            float t = fmaxf(x0, x1);
#pragma unroll
            for (int off = 1; off < 16; off <<= 1)
                t = fmaxf(t, __shfl_xor(t, off, 16));
            float nm    = fmaxf(m_[r], t);
            float alpha = exp2f(m_[r] - nm);
            float p0 = exp2f(x0 - nm);
            float p1 = exp2f(x1 - nm);
            float srow = p0 + p1;
#pragma unroll
            for (int off = 1; off < 16; off <<= 1)
                srow += __shfl_xor(srow, off, 16);
            m_[r] = nm;
            l_[r] = l_[r] * alpha + srow;
#pragma unroll
            for (int nt = 0; nt < 4; nt++) oacc[nt][r] *= alpha;
            int prow = r + 8 * hi;
            P[prow][l15]      = (_Float16)p0;
            P[prow][16 + l15] = (_Float16)p1;
        }
        __syncthreads();

        // --- O += P V  (fragments now contiguous 16B LDS loads)
        v16h pa = cat16(*(const h8*)&P[l15][hi * 8],
                        *(const h8*)&P[l15][16 + hi * 8]);
#pragma unroll
        for (int nt = 0; nt < 4; nt++) {
            int d = nt * 16 + l15;
            v16h vb = cat16(*(const h8*)&VT[d][hi * 8],
                            *(const h8*)&VT[d][16 + hi * 8]);
            oacc[nt] = __builtin_amdgcn_wmma_f32_16x16x32_f16(
                false, pa, false, vb, (short)0, oacc[nt], false, false);
        }
        __syncthreads();
    }

    // --- normalize and store O
#pragma unroll
    for (int r = 0; r < 8; r++) {
        float inv = 1.0f / l_[r];
        int row = qb * 16 + r + 8 * hi;
        size_t base = (size_t)(b * T + row) * E + h * D;
#pragma unroll
        for (int nt = 0; nt < 4; nt++)
            o[base + nt * 16 + l15] = (_Float16)(oacc[nt][r] * inv);
    }
}

// =================================================================
// Host-side orchestration
// =================================================================
extern "C" void kernel_launch(void* const* d_in, const int* in_sizes, int n_in,
                              void* d_out, int out_size, void* d_ws, size_t ws_size,
                              hipStream_t stream) {
    (void)in_sizes; (void)n_in; (void)out_size; (void)ws_size;

    const int*   idx     = (const int*)  d_in[0];
    const float* tok_w   = (const float*)d_in[1];
    const float* pos_emb = (const float*)d_in[2];
    const float* ln1_s   = (const float*)d_in[3];
    const float* ln1_b   = (const float*)d_in[4];
    const float* wq      = (const float*)d_in[5];
    const float* wk      = (const float*)d_in[6];
    const float* wv      = (const float*)d_in[7];
    const float* wo      = (const float*)d_in[8];
    const float* bo      = (const float*)d_in[9];
    const float* ln2_s   = (const float*)d_in[10];
    const float* ln2_b   = (const float*)d_in[11];
    const float* w1      = (const float*)d_in[12];
    const float* b1      = (const float*)d_in[13];
    const float* w2      = (const float*)d_in[14];
    const float* b2      = (const float*)d_in[15];
    const float* lnf_s   = (const float*)d_in[16];
    const float* lnf_b   = (const float*)d_in[17];
    const float* head_w  = (const float*)d_in[18];
    const float* head_b  = (const float*)d_in[19];

    // ---- workspace carve-up
    char* ws = (char*)d_ws;
    size_t off = 0;
    float*    x     = (float*)   (ws + off); off += (size_t)M0 * E  * sizeof(float);
    _Float16* h16   = (_Float16*)(ws + off); off += (size_t)M0 * E  * sizeof(_Float16);
    _Float16* q16   = (_Float16*)(ws + off); off += (size_t)M0 * E  * sizeof(_Float16);
    _Float16* k16   = (_Float16*)(ws + off); off += (size_t)M0 * E  * sizeof(_Float16);
    _Float16* v16b  = (_Float16*)(ws + off); off += (size_t)M0 * E  * sizeof(_Float16);
    _Float16* o16   = (_Float16*)(ws + off); off += (size_t)M0 * E  * sizeof(_Float16);
    _Float16* f16bf = (_Float16*)(ws + off); off += (size_t)M0 * FF * sizeof(_Float16);

    embed_kernel<<<M0, 256, 0, stream>>>(idx, tok_w, pos_emb, x);

    dim3 gE(E / 128, M0 / 128);
    dim3 gF(FF / 128, M0 / 128);
    dim3 gV(V / 128, M0 / 128);
    dim3 gA(T / 16, B * H);

    for (int l = 0; l < L; l++) {
        ln_kernel<<<M0, 256, 0, stream>>>(x, ln1_s + (size_t)l * E,
                                          ln1_b + (size_t)l * E, h16);
        wmma_gemm<<<gE, 256, 0, stream>>>(h16, wq + (size_t)l * E * E,
                                          nullptr, nullptr, nullptr, q16,
                                          M0, E, E, 8);
        wmma_gemm<<<gE, 256, 0, stream>>>(h16, wk + (size_t)l * E * E,
                                          nullptr, nullptr, nullptr, k16,
                                          M0, E, E, 8);
        wmma_gemm<<<gE, 256, 0, stream>>>(h16, wv + (size_t)l * E * E,
                                          nullptr, nullptr, nullptr, v16b,
                                          M0, E, E, 8);
        attn_kernel<<<gA, 32, 0, stream>>>(q16, k16, v16b, o16);
        wmma_gemm<<<gE, 256, 0, stream>>>(o16, wo + (size_t)l * E * E,
                                          bo + (size_t)l * E, x, x, nullptr,
                                          M0, E, E, 6);
        ln_kernel<<<M0, 256, 0, stream>>>(x, ln2_s + (size_t)l * E,
                                          ln2_b + (size_t)l * E, h16);
        wmma_gemm<<<gF, 256, 0, stream>>>(h16, w1 + (size_t)l * E * FF,
                                          b1 + (size_t)l * FF, nullptr,
                                          nullptr, f16bf,
                                          M0, FF, E, 13);
        wmma_gemm<<<gE, 256, 0, stream>>>(f16bf, w2 + (size_t)l * FF * E,
                                          b2 + (size_t)l * E, x, x, nullptr,
                                          M0, E, FF, 6);
    }

    ln_kernel<<<M0, 256, 0, stream>>>(x, lnf_s, lnf_b, h16);
    wmma_gemm<<<gV, 256, 0, stream>>>(h16, head_w, head_b, nullptr,
                                      (float*)d_out, nullptr,
                                      M0, V, E, 4);
}
